// Model_35983236006186
// MI455X (gfx1250) — matrix operand order repaired and checked by reading, not compiled, not run on hardware
//
#include <hip/hip_runtime.h>
#include <hip/hip_bf16.h>

typedef __attribute__((ext_vector_type(16))) _Float16 v16h;
typedef __attribute__((ext_vector_type(8)))  float    v8f;

// ---------------------------------------------------------------------------
// WMMA fragment loaders for V_WMMA_F32_16X16X32_F16 (wave32); A/B in contiguous k order, same for both.
// A (16x32, row-major, stride lda in halves):
//   lane<16: row=lane, VGPR v holds K=2v..2v+1 (K 0..15)
//   lane>=16: same rows, K offset +16 (K 16..31)
// B (32x16) loaded from an N-major ("Bt") buffer: Bt[n*ldb + k]:
//   col=lane&15, K offset 0 (lanes<16) / 16 (lanes>=16), VGPR v = K ko+2v..+1
// C/D (16x16 f32): VGPR r -> row r (lanes<16) / 8+r (lanes>=16), col=lane&15
// ---------------------------------------------------------------------------
__device__ __forceinline__ v16h load_a_frag(const _Float16* A, int lda, int lane) {
  int r  = lane & 15;
  int ko = (lane & 16) ? 16 : 0;
  const _Float16* row = A + r * lda + ko;
  v16h f;
  union { unsigned int u; _Float16 h[2]; } cv;
#pragma unroll
  for (int v = 0; v < 4; ++v) {
    cv.u = *(const unsigned int*)(row + 2 * v);
    f[2 * v] = cv.h[0]; f[2 * v + 1] = cv.h[1];
  }
#pragma unroll
  for (int v = 0; v < 4; ++v) {
    cv.u = *(const unsigned int*)(row + 8 + 2 * v);
    f[8 + 2 * v] = cv.h[0]; f[9 + 2 * v] = cv.h[1];
  }
  return f;
}

__device__ __forceinline__ v16h load_b_frag(const _Float16* Bt, int ldb, int lane) {
  int c  = lane & 15;
  int ko = (lane & 16) ? 16 : 0;
  const _Float16* p = Bt + c * ldb + ko;
  v16h f;
  union { unsigned int u; _Float16 h[2]; } cv;
#pragma unroll
  for (int v = 0; v < 8; ++v) {
    cv.u = *(const unsigned int*)(p + 2 * v);
    f[2 * v] = cv.h[0]; f[2 * v + 1] = cv.h[1];
  }
  return f;
}

// ---------------------------------------------------------------------------
// K1: LabEmbedding. sum_t w_t == 1 when denom>0, so
// lab = (sum t*v / sum t) * lab_W + 1{denom>0} * lab_b
// ---------------------------------------------------------------------------
__global__ __launch_bounds__(256) void lab_kernel(
    const float* __restrict__ mt, const float* __restrict__ mv,
    const int* __restrict__ len, const float* __restrict__ labW,
    const float* __restrict__ labB, float* __restrict__ lab) {
  const int row = blockIdx.x, tid = threadIdx.x;
  const int n = len[row];
  float s1 = 0.f, s2 = 0.f;
  for (int i = tid; i < 2048; i += 256) {
    if (i < n) {
      float t = mt[(size_t)row * 2048 + i];
      s1 += t;
      s2 += t * mv[(size_t)row * 2048 + i];
    }
  }
  __shared__ float r1[256], r2[256];
  r1[tid] = s1; r2[tid] = s2;
  __syncthreads();
  for (int off = 128; off > 0; off >>= 1) {
    if (tid < off) { r1[tid] += r1[tid + off]; r2[tid] += r2[tid + off]; }
    __syncthreads();
  }
  const float denom = r1[0];
  const float wv = denom > 0.f ? r2[0] / denom : 0.f;
  const float sw = denom > 0.f ? 1.f : 0.f;
  if (tid < 128)
    lab[row * 128 + tid] = wv * labW[row * 128 + tid] + sw * labB[row * 128 + tid];
}

// K2: per-head column sums of Wk and bk (rank-1 score trick)
__global__ __launch_bounds__(256) void wksum_kernel(
    const float* __restrict__ Wk, const float* __restrict__ bk,
    float* __restrict__ Wksum, float* __restrict__ bksum) {
  int idx = blockIdx.x * 256 + threadIdx.x;
  if (idx < 768 * 8) {
    int d = idx >> 3, h = idx & 7;
    float s = 0.f;
    for (int e = 0; e < 96; ++e) s += Wk[d * 768 + h * 96 + e];
    Wksum[idx] = s;
  } else if (idx < 768 * 8 + 8) {
    int h = idx - 768 * 8;
    float s = 0.f;
    for (int e = 0; e < 96; ++e) s += bk[h * 96 + e];
    bksum[h] = s;
  }
}

// K3: Ksum[s,h] = src[s,:] . Wksum[:,h] + bksum[h]   (stored h-major: Ks[h*256+s])
__global__ __launch_bounds__(256) void ksum_kernel(
    const float* __restrict__ src, const float* __restrict__ Wksum,
    const float* __restrict__ bksum, float* __restrict__ Ks) {
  int h = blockIdx.x, s = threadIdx.x;
  float acc = bksum[h];
  for (int e = 0; e < 768; ++e) acc += src[s * 768 + e] * Wksum[e * 8 + h];
  Ks[h * 256 + s] = acc;
}

// K4: per-head min/max of Ksum (for stable softmax)
__global__ __launch_bounds__(256) void hminmax_kernel(
    const float* __restrict__ Ks, float* __restrict__ hmin, float* __restrict__ hmax) {
  int h = blockIdx.x, s = threadIdx.x;
  float v = Ks[h * 256 + s];
  __shared__ float mn[256], mx[256];
  mn[s] = v; mx[s] = v;
  __syncthreads();
  for (int off = 128; off > 0; off >>= 1) {
    if (s < off) { mn[s] = fminf(mn[s], mn[s + off]); mx[s] = fmaxf(mx[s], mx[s + off]); }
    __syncthreads();
  }
  if (s == 0) { hmin[h] = mn[0]; hmax[h] = mx[0]; }
}

// K5: V = value_embedding @ Wv + bv -> Vt f16, N-major: Vt[j*256 + s]
__global__ __launch_bounds__(256) void projv_kernel(
    const float* __restrict__ ve, const float* __restrict__ Wv,
    const float* __restrict__ bv, _Float16* __restrict__ Vt) {
  int s = blockIdx.x, tid = threadIdx.x;
  __shared__ float row[768];
  for (int e = tid; e < 768; e += 256) row[e] = ve[s * 768 + e];
  __syncthreads();
  for (int j = tid; j < 768; j += 256) {
    float acc = bv[j];
    for (int e = 0; e < 768; ++e) acc += row[e] * Wv[e * 768 + j];
    Vt[j * 256 + s] = (_Float16)acc;
  }
}

// K6: WoT[n*768+k] = (f16) Wo[k*768+n]
__global__ __launch_bounds__(256) void wot_kernel(
    const float* __restrict__ Wo, _Float16* __restrict__ WoT) {
  int idx = blockIdx.x * 256 + threadIdx.x;
  int n = idx / 768, k = idx % 768;
  WoT[idx] = (_Float16)Wo[k * 768 + n];
}

// ---------------------------------------------------------------------------
// K7 (stage 1): per (series bt, head h): softmax A[64,256] in LDS (f16), then
// WMMA GEMM A[64,256] x Vh[256,96] -> rep f16 tile.
// ---------------------------------------------------------------------------
__global__ __launch_bounds__(256) void stage1_kernel(
    const float* __restrict__ ts, const float* __restrict__ Ks,
    const float* __restrict__ hmin, const float* __restrict__ hmax,
    const _Float16* __restrict__ Vt, _Float16* __restrict__ rep) {
  const int bt = blockIdx.x;        // 0..511 (= b*64 + var)
  const int h  = blockIdx.y;        // 0..7
  const int tid = threadIdx.x, w = tid >> 5, lane = tid & 31;
  const int bb = bt >> 6, nvar = bt & 63;

  __shared__ __align__(16) _Float16 Ash[64][264];
  __shared__ float csh[256];
  __shared__ float cmin_s, cmax_s;

  csh[tid] = Ks[h * 256 + tid];
  if (tid == 0) { cmin_s = hmin[h]; cmax_s = hmax[h]; }
  __syncthreads();

  const float scale = 0.10206207261596575f;   // 1/sqrt(96)
  for (int t = w * 8; t < w * 8 + 8; ++t) {
    float q = ts[(bb * 64 + t) * 64 + nvar];  // time_series[bb, t, nvar]
    if (q != q) q = 0.f;                      // NaN mask
    const float x = scale * q;
    const float m = (x >= 0.f) ? x * cmax_s : x * cmin_s;
    float Z = 0.f;
    for (int s = lane; s < 256; s += 32) Z += __expf(x * csh[s] - m);
    for (int off = 16; off > 0; off >>= 1) Z += __shfl_xor(Z, off, 32);
    const float iZ = 1.f / Z;                 // max attained => Z >= 1
    for (int s = lane; s < 256; s += 32)
      Ash[t][s] = (_Float16)(__expf(x * csh[s] - m) * iZ);
  }
  __syncthreads();

  // 4x6 = 24 output tiles of 16x16; 3 per wave; K = 256 -> 8 wmma steps
  for (int idx = w; idx < 24; idx += 8) {
    const int m0 = (idx / 6) * 16, n0 = (idx % 6) * 16;
    v8f acc = {};
#pragma unroll
    for (int kk = 0; kk < 8; ++kk) {
      v16h a = load_a_frag(&Ash[m0][kk * 32], 264, lane);
      v16h b = load_b_frag(Vt + (h * 96 + n0) * 256 + kk * 32, 256, lane);
      acc = __builtin_amdgcn_wmma_f32_16x16x32_f16(false, a, false, b,
                                                   (short)0, acc, false, false);
    }
    const int col = lane & 15, rb = (lane & 16) ? 8 : 0;
#pragma unroll
    for (int r = 0; r < 8; ++r)
      rep[(size_t)(bt * 64 + m0 + rb + r) * 768 + h * 96 + n0 + col] = (_Float16)acc[r];
  }
}

// ---------------------------------------------------------------------------
// K8 (stage 2): reprog_out = rep[32768,768] @ Wo[768,768] + bo, fp32 out.
// 64x96 macro-tile per workgroup; WoT (1.2MB f16) is L2-resident.
// ---------------------------------------------------------------------------
__global__ __launch_bounds__(256) void stage2_kernel(
    const _Float16* __restrict__ rep, const _Float16* __restrict__ WoT,
    const float* __restrict__ bo, float* __restrict__ out) {
  const int r0 = blockIdx.x * 64;   // 512 blocks
  const int c0 = blockIdx.y * 96;   // 8 blocks
  const int tid = threadIdx.x, w = tid >> 5, lane = tid & 31;
  for (int idx = w; idx < 24; idx += 8) {
    const int m0 = (idx / 6) * 16, n0 = (idx % 6) * 16;
    const _Float16* Abase = rep + (size_t)(r0 + m0) * 768;
    const _Float16* Bbase = WoT + (size_t)(c0 + n0) * 768;
    v8f acc = {};
#pragma unroll 4
    for (int kk = 0; kk < 24; ++kk) {
      v16h a = load_a_frag(Abase + kk * 32, 768, lane);
      v16h b = load_b_frag(Bbase + kk * 32, 768, lane);
      acc = __builtin_amdgcn_wmma_f32_16x16x32_f16(false, a, false, b,
                                                   (short)0, acc, false, false);
    }
    const int col = lane & 15, rb = (lane & 16) ? 8 : 0;
    const float bias = bo[c0 + n0 + col];
#pragma unroll
    for (int r = 0; r < 8; ++r)
      out[(size_t)(r0 + m0 + rb + r) * 768 + c0 + n0 + col] = acc[r] + bias;
  }
}

// K9: variable-attention Q/K/V projections (fp32, tiny)
__global__ __launch_bounds__(256) void qkv_va_kernel(
    const float* __restrict__ lab,
    const float* __restrict__ Wq, const float* __restrict__ bq,
    const float* __restrict__ Wk, const float* __restrict__ bk,
    const float* __restrict__ Wv, const float* __restrict__ bv,
    float* __restrict__ Q, float* __restrict__ K, float* __restrict__ V) {
  int idx = blockIdx.x * 256 + threadIdx.x;      // bn*1024 + j
  int bn = idx >> 10, j = idx & 1023;
  float aq = bq[j], ak = bk[j], av = bv[j];
  const float* lrow = lab + bn * 128;
  for (int e = 0; e < 128; ++e) {
    float lv = lrow[e];
    aq += lv * Wq[e * 1024 + j];
    ak += lv * Wk[e * 1024 + j];
    av += lv * Wv[e * 1024 + j];
  }
  Q[idx] = aq; K[idx] = ak; V[idx] = av;
}

// K10: collapsed variable attention per (b,h):
// kw = sum_m Ws[m]*K_m; summ[n] = Q_n.kw/sqrt(128)+bs; beta=softmax_n; res=sum beta*V
__global__ __launch_bounds__(128) void va_attn_kernel(
    const float* __restrict__ Q, const float* __restrict__ K,
    const float* __restrict__ V, const float* __restrict__ Ws,
    const float* __restrict__ bs, float* __restrict__ res) {
  const int b = blockIdx.x, h = blockIdx.y, tid = threadIdx.x;
  __shared__ float kw[128], summ[64], beta[64];
  {
    float s = 0.f;
    for (int m = 0; m < 64; ++m)
      s += Ws[m] * K[(size_t)(b * 64 + m) * 1024 + h * 128 + tid];
    kw[tid] = s;
  }
  __syncthreads();
  if (tid < 64) {
    const float* qrow = Q + (size_t)(b * 64 + tid) * 1024 + h * 128;
    float s = 0.f;
    for (int d = 0; d < 128; ++d) s += qrow[d] * kw[d];
    summ[tid] = s * 0.08838834764831845f + bs[0];   // 1/sqrt(128)
  }
  __syncthreads();
  if (tid == 0) {
    float mx = summ[0];
    for (int n = 1; n < 64; ++n) mx = fmaxf(mx, summ[n]);
    float Z = 0.f;
    for (int n = 0; n < 64; ++n) { beta[n] = __expf(summ[n] - mx); Z += beta[n]; }
    float iZ = 1.f / Z;
    for (int n = 0; n < 64; ++n) beta[n] *= iZ;
  }
  __syncthreads();
  {
    float s = 0.f;
    for (int n = 0; n < 64; ++n)
      s += beta[n] * V[(size_t)(b * 64 + n) * 1024 + h * 128 + tid];
    res[(b * 8 + h) * 128 + tid] = s;               // res[b, h*128+d]
  }
}

// K11: out_projection + classification head -> logits[16]
__global__ __launch_bounds__(256) void head_kernel(
    const float* __restrict__ res, const float* __restrict__ Wova,
    const float* __restrict__ bova, const float* __restrict__ W1,
    const float* __restrict__ b1, const float* __restrict__ W2,
    const float* __restrict__ b2, float* __restrict__ logits) {
  __shared__ float va[8][128], h1[8][128];
  const int tid = threadIdx.x;
  for (int idx = tid; idx < 1024; idx += 256) {
    int b = idx >> 7, j = idx & 127;
    float s = bova[j];
    const float* rrow = res + b * 1024;
    for (int e = 0; e < 1024; ++e) s += rrow[e] * Wova[e * 128 + j];
    va[b][j] = s;
  }
  __syncthreads();
  for (int idx = tid; idx < 1024; idx += 256) {
    int b = idx >> 7, j = idx & 127;
    float s = b1[j];
    for (int e = 0; e < 128; ++e) s += va[b][e] * W1[e * 128 + j];
    h1[b][j] = fmaxf(s, 0.f);
  }
  __syncthreads();
  if (tid < 16) {
    int b = tid >> 1, wc = tid & 1;
    float s = b2[wc];
    for (int e = 0; e < 128; ++e) s += h1[b][e] * W2[e * 2 + wc];
    logits[b * 2 + wc] = s;
  }
}

extern "C" void kernel_launch(void* const* d_in, const int* in_sizes, int n_in,
                              void* d_out, int out_size, void* d_ws, size_t ws_size,
                              hipStream_t stream) {
  const float* mt    = (const float*)d_in[0];
  const float* mv    = (const float*)d_in[1];
  const int*   len   = (const int*)  d_in[2];
  const float* ts    = (const float*)d_in[3];
  const float* src   = (const float*)d_in[4];
  const float* ve    = (const float*)d_in[5];
  const float* labW  = (const float*)d_in[6];
  const float* labB  = (const float*)d_in[7];
  const float* Wk    = (const float*)d_in[8];
  const float* bk    = (const float*)d_in[9];
  const float* Wv    = (const float*)d_in[10];
  const float* bv    = (const float*)d_in[11];
  const float* Wo    = (const float*)d_in[12];
  const float* bo    = (const float*)d_in[13];
  const float* Wq_va = (const float*)d_in[14];
  const float* bq_va = (const float*)d_in[15];
  const float* Wk_va = (const float*)d_in[16];
  const float* bk_va = (const float*)d_in[17];
  const float* Wv_va = (const float*)d_in[18];
  const float* bv_va = (const float*)d_in[19];
  const float* Ws    = (const float*)d_in[20];
  const float* bs    = (const float*)d_in[21];
  const float* Wova  = (const float*)d_in[22];
  const float* bova  = (const float*)d_in[23];
  const float* W1    = (const float*)d_in[24];
  const float* b1    = (const float*)d_in[25];
  const float* W2    = (const float*)d_in[26];
  const float* b2    = (const float*)d_in[27];

  char* ws = (char*)d_ws;
  size_t off = 0;
  auto alloc = [&](size_t bytes) -> void* {
    void* p = ws + off;
    off = (off + bytes + 255) & ~(size_t)255;
    return p;
  };
  float*    lab   = (float*)   alloc(512 * 128 * 4);
  float*    Wksum = (float*)   alloc(768 * 8 * 4);
  float*    bksum = (float*)   alloc(8 * 4);
  float*    Ks    = (float*)   alloc(8 * 256 * 4);
  float*    hmin  = (float*)   alloc(8 * 4);
  float*    hmax  = (float*)   alloc(8 * 4);
  _Float16* Vt    = (_Float16*)alloc((size_t)768 * 256 * 2);
  _Float16* WoT   = (_Float16*)alloc((size_t)768 * 768 * 2);
  _Float16* rep   = (_Float16*)alloc((size_t)32768 * 768 * 2);
  float*    Qva   = (float*)   alloc((size_t)512 * 1024 * 4);
  float*    Kva   = (float*)   alloc((size_t)512 * 1024 * 4);
  float*    Vva   = (float*)   alloc((size_t)512 * 1024 * 4);
  float*    resv  = (float*)   alloc(8 * 1024 * 4);

  float* logits = (float*)d_out;        // outputs concatenated: logits[16], reprog
  float* outrep = (float*)d_out + 16;

  lab_kernel   <<<512, 256, 0, stream>>>(mt, mv, len, labW, labB, lab);
  wksum_kernel <<<25, 256, 0, stream>>>(Wk, bk, Wksum, bksum);
  ksum_kernel  <<<8, 256, 0, stream>>>(src, Wksum, bksum, Ks);
  hminmax_kernel<<<8, 256, 0, stream>>>(Ks, hmin, hmax);
  projv_kernel <<<256, 256, 0, stream>>>(ve, Wv, bv, Vt);
  wot_kernel   <<<2304, 256, 0, stream>>>(Wo, WoT);
  stage1_kernel<<<dim3(512, 8), 256, 0, stream>>>(ts, Ks, hmin, hmax, Vt, rep);
  stage2_kernel<<<dim3(512, 8), 256, 0, stream>>>(rep, WoT, bo, outrep);
  qkv_va_kernel<<<2048, 256, 0, stream>>>(lab, Wq_va, bq_va, Wk_va, bk_va,
                                          Wv_va, bv_va, Qva, Kva, Vva);
  va_attn_kernel<<<dim3(8, 8), 128, 0, stream>>>(Qva, Kva, Vva, Ws, bs, resv);
  head_kernel  <<<1, 256, 0, stream>>>(resv, Wova, bova, W1, b1, W2, b2, logits);
}
